// ResidualGNN_19215683683065
// MI455X (gfx1250) — matrix operand order repaired and checked by reading, not compiled, not run on hardware
//
#include <hip/hip_runtime.h>

// CDNA5 / gfx1250: wave32, WMMA 16x16x32 f16 -> f32
typedef __attribute__((ext_vector_type(16))) _Float16 v16h;
typedef __attribute__((ext_vector_type(8)))  _Float16 v8h;
typedef __attribute__((ext_vector_type(2)))  _Float16 v2h;
typedef __attribute__((ext_vector_type(2)))  __fp16   v2fp;   // return type of cvt_pkrtz
typedef __attribute__((ext_vector_type(8)))  float    v8f;

#define N_PART  1024
#define B_BATCH 8
#define HID     32
#define HP      24          // halves per column of transposed h-scratch (16 + 8 pad): 48B pitch,
                            // keeps every 8-half row-group 16B-aligned; dword stride 12 -> conflict-free
#define NWAVES  8
#define WAVE_H  (2 * HID * HP)   // two scratch buffers (IJ / JI) per wave, in halves

__device__ __forceinline__ float silu_f(float x) {
    // x*sigmoid(x) = 0.5x*(1 + tanh(0.5x)) -> v_mul + v_tanh_f32 + v_fma (gfx1250 tanh trans op)
    float h = 0.5f * x;
    return __builtin_amdgcn_tanhf(h) * h + h;
}

__device__ __forceinline__ v8h pack_silu(v8f c) {
    v8h hv;
    #pragma unroll
    for (int g = 0; g < 8; g += 2) {
        v2fp pk = __builtin_amdgcn_cvt_pkrtz(silu_f(c[g]), silu_f(c[g + 1]));
        v2h  p  = __builtin_bit_cast(v2h, pk);
        hv[g] = p[0]; hv[g + 1] = p[1];
    }
    return hv;
}

struct LaneWeights {
    v16h w1a, w1b;   // W1 B-tiles (K padded 13->32, N cols 0..15 / 16..31)
    v16h w2a, w2b;   // W2 B-tiles (K=32 exact)
    v16h w3;         // W3 B-tile  (N cols 0..2 valid, rest zero)
    float b1lo, b1hi, b2lo, b2hi;
};

__global__ void __launch_bounds__(256)
gnn_edge_kernel(const float* __restrict__ r, const float* __restrict__ v,
                const float* __restrict__ W1, const float* __restrict__ b1,
                const float* __restrict__ W2, const float* __restrict__ b2,
                const float* __restrict__ W3, float* __restrict__ out)
{
    __shared__ float sr[N_PART * 3];
    __shared__ float sv[N_PART * 3];
    __shared__ __align__(16) _Float16 hbuf[NWAVES * WAVE_H];   // transposed h scratch: [col][row]
    __shared__ float sres[NWAVES * 3];
    __shared__ float sphys[NWAVES * 3];

    const int tid  = threadIdx.x;
    const int lane = tid & 31;
    const int wave = tid >> 5;
    const int b    = blockIdx.x / N_PART;
    const int i    = blockIdx.x % N_PART;

    // stage r,v for this batch into LDS (reused by all j-tiles of this block)
    const float* rb = r + (size_t)b * N_PART * 3;
    const float* vb = v + (size_t)b * N_PART * 3;
    __builtin_prefetch(rb, 0, 0);
    __builtin_prefetch(vb, 0, 0);
    for (int t = tid; t < N_PART * 3; t += 256) {
        sr[t] = rb[t];
        sv[t] = vb[t];
    }

    // per-lane weight B-tiles: element e of lane l holds W[k][n], k = (e<8 ? e : 8+e) + (l<16 ? 0 : 8), n = l&15
    const bool lo = lane < 16;
    const int  n  = lane & 15;
    LaneWeights Wt;
    #pragma unroll
    for (int e = 0; e < 16; ++e) {
        int k = (e < 8 ? e : 8 + e) + (lo ? 0 : 8);
        Wt.w1a[e] = (k < 13) ? (_Float16)W1[k * HID + n]      : (_Float16)0.0f;
        Wt.w1b[e] = (k < 13) ? (_Float16)W1[k * HID + 16 + n] : (_Float16)0.0f;
        Wt.w2a[e] = (_Float16)W2[k * HID + n];
        Wt.w2b[e] = (_Float16)W2[k * HID + 16 + n];
        Wt.w3[e]  = (n < 3) ? (_Float16)W3[k * 3 + n] : (_Float16)0.0f;
    }
    Wt.b1lo = b1[n]; Wt.b1hi = b1[16 + n];
    Wt.b2lo = b2[n]; Wt.b2hi = b2[16 + n];

    __syncthreads();

    const float rix = sr[i * 3 + 0], riy = sr[i * 3 + 1], riz = sr[i * 3 + 2];
    const float vix = sv[i * 3 + 0], viy = sv[i * 3 + 1], viz = sv[i * 3 + 2];

    _Float16* hA = hbuf + wave * WAVE_H;        // IJ scratch
    _Float16* hB = hA + HID * HP;               // JI scratch (separate -> passes can interleave)
    const int rbase = lo ? 0 : 8;               // D-tile rows held by this lane half

    float accRes = 0.0f;                        // per-lane partial of a_res (column n)
    float apx = 0.0f, apy = 0.0f, apz = 0.0f;   // per-lane partial of a_phys (lanes<16 only)

    #pragma unroll 1
    for (int it = 0; it < N_PART / (16 * NWAVES); ++it) {
        const int j = it * (16 * NWAVES) + wave * 16 + n;
        const float rjx = sr[j * 3 + 0], rjy = sr[j * 3 + 1], rjz = sr[j * 3 + 2];
        const float vjx = sv[j * 3 + 0], vjy = sv[j * 3 + 1], vjz = sv[j * 3 + 2];
        const float dxx = rjx - rix, dxy = rjy - riy, dxz = rjz - riz;
        const float dvx = vjx - vix, dvy = vjy - viy, dvz = vjz - viz;
        const float d2   = dxx * dxx + dxy * dxy + dxz * dxz;
        const float dist = __builtin_amdgcn_sqrtf(d2 + 1e-8f);

        // a_phys: dx * (d2 + EPS^2)^-1.5 via rsq^3; diagonal contributes 0 (dx == 0)
        const float wv  = d2 + 0.0001f;
        const float s15 = __builtin_amdgcn_rsqf(wv);
        const float inv = s15 * s15 * s15;
        if (lo) { apx += dxx * inv; apy += dxy * inv; apz += dxz * inv; }

        // A-tiles. IJ features [v_i, v_j, dx, dv, dist]; lanes<16 -> k 0..7, lanes>=16 -> k 8..12
        v16h aIJ = {}, aJI = {};
        aIJ[0] = (_Float16)(lo ? vix : dxz);
        aIJ[1] = (_Float16)(lo ? viy : dvx);
        aIJ[2] = (_Float16)(lo ? viz : dvy);
        aIJ[3] = (_Float16)(lo ? vjx : dvz);
        aIJ[4] = (_Float16)(lo ? vjy : dist);
        aIJ[5] = (_Float16)(lo ? vjz : 0.0f);
        aIJ[6] = (_Float16)(lo ? dxx : 0.0f);
        aIJ[7] = (_Float16)(lo ? dxy : 0.0f);
        // JI features [v_j, v_i, -dx, -dv, dist]
        aJI[0] = (_Float16)(lo ? vjx : -dxz);
        aJI[1] = (_Float16)(lo ? vjy : -dvx);
        aJI[2] = (_Float16)(lo ? vjz : -dvy);
        aJI[3] = (_Float16)(lo ? vix : -dvz);
        aJI[4] = (_Float16)(lo ? viy : dist);
        aJI[5] = (_Float16)(lo ? viz : 0.0f);
        aJI[6] = (_Float16)(lo ? -dxx : 0.0f);
        aJI[7] = (_Float16)(lo ? -dxy : 0.0f);

        // ---- layer 1 (both directions, 4 independent WMMAs) ----
        v8f cI0, cI1, cJ0, cJ1;
        #pragma unroll
        for (int g = 0; g < 8; ++g) {
            cI0[g] = Wt.b1lo; cI1[g] = Wt.b1hi;
            cJ0[g] = Wt.b1lo; cJ1[g] = Wt.b1hi;
        }
        cI0 = __builtin_amdgcn_wmma_f32_16x16x32_f16(false, aIJ, false, Wt.w1a, (short)0, cI0, false, false);
        cI1 = __builtin_amdgcn_wmma_f32_16x16x32_f16(false, aIJ, false, Wt.w1b, (short)0, cI1, false, false);
        cJ0 = __builtin_amdgcn_wmma_f32_16x16x32_f16(false, aJI, false, Wt.w1a, (short)0, cJ0, false, false);
        cJ1 = __builtin_amdgcn_wmma_f32_16x16x32_f16(false, aJI, false, Wt.w1b, (short)0, cJ1, false, false);

        // silu -> f16, one b128 store per lane per tile (transposed layout: col-major, 8 rows contiguous)
        *(v8h*)(hA + n * HP + rbase)          = pack_silu(cI0);
        *(v8h*)(hA + (16 + n) * HP + rbase)   = pack_silu(cI1);
        *(v8h*)(hB + n * HP + rbase)          = pack_silu(cJ0);
        *(v8h*)(hB + (16 + n) * HP + rbase)   = pack_silu(cJ1);

        // gather back in A layout: a[e] = h[row = n][k(e,lane)] = hT[k*HP + n]
        v16h a1I, a1J;
        #pragma unroll
        for (int e = 0; e < 16; ++e) {
            int k = e + (e < 8 ? 0 : 8) + (lo ? 0 : 8);
            a1I[e] = hA[k * HP + n];
            a1J[e] = hB[k * HP + n];
        }

        // ---- layer 2 ----
        v8f dI0, dI1, dJ0, dJ1;
        #pragma unroll
        for (int g = 0; g < 8; ++g) {
            dI0[g] = Wt.b2lo; dI1[g] = Wt.b2hi;
            dJ0[g] = Wt.b2lo; dJ1[g] = Wt.b2hi;
        }
        dI0 = __builtin_amdgcn_wmma_f32_16x16x32_f16(false, a1I, false, Wt.w2a, (short)0, dI0, false, false);
        dI1 = __builtin_amdgcn_wmma_f32_16x16x32_f16(false, a1I, false, Wt.w2b, (short)0, dI1, false, false);
        dJ0 = __builtin_amdgcn_wmma_f32_16x16x32_f16(false, a1J, false, Wt.w2a, (short)0, dJ0, false, false);
        dJ1 = __builtin_amdgcn_wmma_f32_16x16x32_f16(false, a1J, false, Wt.w2b, (short)0, dJ1, false, false);

        *(v8h*)(hA + n * HP + rbase)          = pack_silu(dI0);
        *(v8h*)(hA + (16 + n) * HP + rbase)   = pack_silu(dI1);
        *(v8h*)(hB + n * HP + rbase)          = pack_silu(dJ0);
        *(v8h*)(hB + (16 + n) * HP + rbase)   = pack_silu(dJ1);

        v16h a2I, a2J;
        #pragma unroll
        for (int e = 0; e < 16; ++e) {
            int k = e + (e < 8 ? 0 : 8) + (lo ? 0 : 8);
            a2I[e] = hA[k * HP + n];
            a2J[e] = hB[k * HP + n];
        }

        // ---- layer 3: 32 -> 3; b3 cancels in m_ij - m_ji ----
        v8f e3I = {}, e3J = {};
        e3I = __builtin_amdgcn_wmma_f32_16x16x32_f16(false, a2I, false, Wt.w3, (short)0, e3I, false, false);
        e3J = __builtin_amdgcn_wmma_f32_16x16x32_f16(false, a2J, false, Wt.w3, (short)0, e3J, false, false);

        float sd = 0.0f;
        #pragma unroll
        for (int g = 0; g < 8; ++g) sd += e3I[g] - e3J[g];   // rows of +m_ij and -m_ji
        accRes += sd;
    }

    // ---- cross-lane reductions (wave32 butterflies), then tiny LDS gather ----
    accRes += __shfl_xor(accRes, 16);            // combine column n held in lanes n and 16+n
    #pragma unroll
    for (int off = 16; off >= 1; off >>= 1) {
        apx += __shfl_xor(apx, off);
        apy += __shfl_xor(apy, off);
        apz += __shfl_xor(apz, off);
    }
    if (lane == 0) {
        sphys[wave * 3 + 0] = apx;
        sphys[wave * 3 + 1] = apy;
        sphys[wave * 3 + 2] = apz;
    }
    if (lo && n < 3) sres[wave * 3 + n] = accRes;
    __syncthreads();

    if (tid < 3) {   // tid = output component
        float tot = 0.0f;
        #pragma unroll
        for (int w8 = 0; w8 < NWAVES; ++w8)
            tot += sres[w8 * 3 + tid] + sphys[w8 * 3 + tid];
        out[((size_t)b * N_PART + i) * 3 + tid] = tot;
    }
}

extern "C" void kernel_launch(void* const* d_in, const int* in_sizes, int n_in,
                              void* d_out, int out_size, void* d_ws, size_t ws_size,
                              hipStream_t stream) {
    const float* r  = (const float*)d_in[0];
    const float* v  = (const float*)d_in[1];
    const float* W1 = (const float*)d_in[2];
    const float* b1 = (const float*)d_in[3];
    const float* W2 = (const float*)d_in[4];
    const float* b2 = (const float*)d_in[5];
    const float* W3 = (const float*)d_in[6];
    // d_in[7] (b3) is unused: it cancels exactly in f_ij = m_ij - m_ji.
    float* out = (float*)d_out;

    gnn_edge_kernel<<<dim3(B_BATCH * N_PART), dim3(256), 0, stream>>>(
        r, v, W1, b1, W2, b2, W3, out);
}
